// SparseMemory_25383256719711
// MI455X (gfx1250) — 8-bit matrix operand order repaired and checked by reading, compiled, not run on hardware
//
#include <hip/hip_runtime.h>
#include <hip/hip_bf16.h>


#define Bdim 1024
#define Tdim 1024
#define Ndim 2048
#define NBIT 16
#define Mdim 65536

typedef __attribute__((ext_vector_type(8))) int v8i;


__global__ void pack_bits_T_kernel(const int* __restrict__ in,
                                   unsigned int* __restrict__ aPack) {
    const int idx = blockIdx.x * blockDim.x + threadIdx.x;
    const int tq = idx >> 10;
    const int b  = idx & 1023;
    const int* p = in + (size_t)b * Tdim + (tq << 2);
    const unsigned int b0 = (p[0] != 0) ? 1u : 0u;
    const unsigned int b1 = (p[1] != 0) ? 1u : 0u;
    const unsigned int b2 = (p[2] != 0) ? 1u : 0u;
    const unsigned int b3 = (p[3] != 0) ? 1u : 0u;
    aPack[idx] = b0 | (b1 << 8) | (b2 << 16) | (b3 << 24);
}

__global__ void zero_u32_kernel(unsigned int* __restrict__ w, int nWords) {
    int i = blockIdx.x * blockDim.x + threadIdx.x;
    if (i < nWords) w[i] = 0u;
}

__global__ void scatter_w_kernel(const int* __restrict__ conn,
                                 unsigned char* __restrict__ Wlo8,
                                 unsigned char* __restrict__ Whi8) {
    int n = blockIdx.x * blockDim.x + threadIdx.x;
    if (n >= Ndim) return;
    for (int j = 0; j < NBIT; ++j) {
        const int t = conn[n * NBIT + j];
        const size_t a = ((size_t)(t >> 2) * Ndim + n) * 4 + (t & 3);
        if (j < 8) Whi8[a] = (unsigned char)(Whi8[a] + (1u << (7 - j)));
        else       Wlo8[a] = (unsigned char)(Wlo8[a] + (1u << (15 - j)));
    }
}


__device__ __forceinline__ void load_frags(
    const unsigned int* __restrict__ aP,
    const unsigned int* __restrict__ lo,
    const unsigned int* __restrict__ hi,
    int b0, int nCol, int m, int half, int kq0,
    v8i& fa, v8i& flo, v8i& fhi)
{
#pragma unroll
    for (int v = 0; v < 8; ++v) {
        const int kqa = ((v >> 1) << 2) + (v & 1) + (half << 1);
        fa[v] = (int)aP[(size_t)(kq0 + kqa) * Bdim + b0 + m];
        const int kqb = ((v >> 1) << 2) + (v & 1) + (half << 1);
        const size_t bi = (size_t)(kq0 + kqb) * Ndim + nCol;
        flo[v] = (int)lo[bi];
        fhi[v] = (int)hi[bi];
    }
}

__global__ __launch_bounds__(256) void wnn_gemm_lookup(
    const unsigned int* __restrict__ aPack,
    const unsigned int* __restrict__ Wlo32,
    const unsigned int* __restrict__ Whi32,
    const float* __restrict__ memory_table,
    float* __restrict__ out)
{
    const int tid  = threadIdx.x;
    const int lane = tid & 31;
    const int wid  = blockIdx.x * 8 + (tid >> 5);

    const int b0 = (wid & 63) << 4;
    const int n0 = (wid >> 6) << 4;

    const int m    = lane & 15;
    const int half = lane >> 4;
    const int nCol = n0 + (lane & 15);

    v8i accLo = {0,0,0,0,0,0,0,0};
    v8i accHi = {0,0,0,0,0,0,0,0};

    v8i fa, flo, fhi, na, nlo, nhi;
    load_frags(aPack, Wlo32, Whi32, b0, nCol, m, half, 0, fa, flo, fhi);

#pragma unroll
    for (int step = 0; step < 16; ++step) {
        if (step < 15)
            load_frags(aPack, Wlo32, Whi32, b0, nCol, m, half,
                       (step + 1) << 4, na, nlo, nhi);
        accLo = __builtin_amdgcn_wmma_i32_16x16x64_iu8(
            false, fa, false, flo, accLo, false, false);
        accHi = __builtin_amdgcn_wmma_i32_16x16x64_iu8(
            false, fa, false, fhi, accHi, false, false);
        fa = na; flo = nlo; fhi = nhi;
    }

    const float* tabRow = memory_table + (size_t)nCol * Mdim;
#pragma unroll
    for (int r = 0; r < 8; ++r) {
        const int addr = (accLo[r] + (accHi[r] << 8)) & 0xFFFF;
        const int bRow = b0 + r + (half << 3);
        const float v = tabRow[addr];
        out[(size_t)bRow * Ndim + nCol] = (v == 1.0f) ? 1.0f : 0.0f;
    }
}


extern "C" void kernel_launch(void* const* d_in, const int* in_sizes, int n_in,
                              void* d_out, int out_size, void* d_ws, size_t ws_size,
                              hipStream_t stream) {
    const int*   input_bits   = (const int*)d_in[0];
    const int*   connections  = (const int*)d_in[1];
    const float* memory_table = (const float*)d_in[2];
    float*       out          = (float*)d_out;

    unsigned int*  aPack = (unsigned int*)d_ws;
    unsigned char* Wlo8  = (unsigned char*)d_ws + (1u << 20);
    unsigned char* Whi8  = (unsigned char*)d_ws + (3u << 20);

    pack_bits_T_kernel<<<(Tdim / 4) * Bdim / 256, 256, 0, stream>>>(
        input_bits, aPack);
    zero_u32_kernel<<<(1u << 20) / 256, 256, 0, stream>>>(
        (unsigned int*)Wlo8, 1 << 20);
    scatter_w_kernel<<<(Ndim + 255) / 256, 256, 0, stream>>>(
        connections, Wlo8, Whi8);

    wnn_gemm_lookup<<<8192 / 8, 256, 0, stream>>>(
        aPack, (const unsigned int*)Wlo8, (const unsigned int*)Whi8,
        memory_table, out);
}
